// summary_topic_final_aspect_3_21148418965772
// MI455X (gfx1250) — matrix operand order repaired and checked by reading, not compiled, not run on hardware
//
#include <hip/hip_runtime.h>
#include <hip/hip_bf16.h>

// ---------------------------------------------------------------------------
// CDNA5 (gfx1250) implementation.
//
// Model sizes (fixed by the reference):
//   B=128, D=128, UR=IR=10, RL=200, SL=30, A=5, V=50000, NU=NI=10000,
//   K=3 (conv taps), ID=32, P=3.
//
// Key algebraic simplification: gumbel_hard()'s *forward value* is an exact
// one-hot, so einsum('bu,buld->bld', p_u, u_rev) is a gather of ONE review
// per (batch, p). We therefore only run the expensive conv on the selected
// reviews (B*P per side) instead of all B*UR reviews.  The gumbel noise
// itself (JAX threefry) is approximated by plain argmax of the logits.
//
// Conv-as-GEMM: y[l,:] = ReLU( Xi[l,:] @ Wf + bias ),  Xi = im2col [L, 384],
// Wf = W reshaped [3*128, 128] (already contiguous in that layout).
// Executed with v_wmma_f32_16x16x32_f16 on f16 data staged in LDS.
// ---------------------------------------------------------------------------

typedef __attribute__((ext_vector_type(16))) _Float16 v16h;
typedef __attribute__((ext_vector_type(8)))  float    v8f;

#define NB 128   // batch
#define ND 128   // feature dim
#define NR 10    // reviews per user/item
#define NRL 200  // review length
#define NSL 30   // summary length
#define NA 5     // aspects
#define NP 3     // co-attention heads
#define NID 32   // id embedding dim
#define KDIM 384 // 3 * D (im2col K)

// ---------------------------------------------------------------------------
// Pack conv weights W (f32, [3,128,128] == Wf[384,128]) into a contiguous-k
// wave32 WMMA B-fragment layout so a lane's 16 f16 values are contiguous:
//   Wpack[((nt*12 + ki)*32 + lane)*16 + j]
//   n    = nt*16 + (lane & 15), half = lane>>4
//   krow = ki*32 + half*16 + 2*(j>>1) + (j&1)
//   value = Wf[krow, n]
// ---------------------------------------------------------------------------
__global__ void pack_weights(const float* __restrict__ Wu,
                             const float* __restrict__ Wi,
                             _Float16* __restrict__ Pu,
                             _Float16* __restrict__ Pi) {
  int t = blockIdx.x * blockDim.x + threadIdx.x;  // 0 .. 2*49152-1
  const int NELEM = 8 * 12 * 32 * 16;             // 49152 == 384*128
  if (t >= 2 * NELEM) return;
  int which = t >= NELEM;
  int i = t - which * NELEM;
  int j    = i & 15;
  int lane = (i >> 4) & 31;
  int rest = i >> 9;          // nt*12 + ki
  int ki   = rest % 12;
  int nt   = rest / 12;
  int n    = nt * 16 + (lane & 15);
  int half = lane >> 4;
  int krow = ki * 32 + half * 16 + ((j >> 1) << 1) + (j & 1);
  const float* W = which ? Wi : Wu;
  _Float16*    P = which ? Pi : Pu;
  P[i] = (_Float16)W[krow * ND + n];
}

// ---------------------------------------------------------------------------
// Conv + ReLU (+ optional mean over L) via WMMA.
// One block (256 threads = 8 waves) per review/summary instance.
//   MEAN=true : grid = B*R, tokens flat [B*R, L], out = [B*R, 128] (mean)
//   MEAN=false: grid = B,   r = sel[b],  tokens [B, R, L], out = [B, L, 128]
// ---------------------------------------------------------------------------
template <int L, int MT, bool MEAN>
__global__ __launch_bounds__(256)
void conv_wmma(const int* __restrict__ tokens,
               const int* __restrict__ sel, int R,
               const float* __restrict__ emb,
               const _Float16* __restrict__ Wpack,
               const float* __restrict__ bias,
               float* __restrict__ out) {
  constexpr int ROWS = MT * 16 + 2;           // zero pad row 0 and rows >= L+1
  __shared__ _Float16 Xp[ROWS * ND];

  const int tid = threadIdx.x;
  long tokBase, outBase;
  if constexpr (MEAN) {
    tokBase = (long)blockIdx.x * L;
    outBase = (long)blockIdx.x * ND;
  } else {
    int b = blockIdx.x;
    int r = sel[b];
    tokBase = ((long)b * R + r) * L;
    outBase = (long)b * L * ND;
  }

  // Zero the padding rows (row 0 and rows L+1 .. ROWS-1).
  for (int i = tid; i < (ROWS - L) * ND; i += 256) {
    int zi = i >> 7, d = i & 127;
    int row = (zi == 0) ? 0 : (L + zi);
    Xp[row * ND + d] = (_Float16)0.0f;
  }
  // Gather embedding rows, convert f32 -> f16 into LDS (shifted by +1 row).
  for (int i = tid; i < L * ND; i += 256) {
    int l = i >> 7, d = i & 127;
    int idx = tokens[tokBase + l];
    Xp[(l + 1) * ND + d] = (_Float16)emb[(long)idx * ND + d];
  }
  __syncthreads();

  const int wv   = tid >> 5;        // wave id: owns N strip [wv*16, wv*16+16)
  const int lane = tid & 31;
  const int m    = lane & 15;
  const int half = lane >> 4;
  const int n    = wv * 16 + m;     // output column for C/D fragments
  const float bn = bias[n];
  float acc = 0.0f;                 // MEAN-mode column accumulator

  for (int mt = 0; mt < MT; ++mt) {
    v8f c = {};
#pragma unroll
    for (int ki = 0; ki < 12; ++ki) {
      // B fragment: one contiguous 32B load per lane (pre-packed layout).
      v16h bf = *reinterpret_cast<const v16h*>(
          Wpack + (((wv * 12 + ki) * 32 + lane) << 4));
      // A fragment: contiguous k = 16*half + s, same order as Wpack. im2col column c = ki*32 + off,
      // shift k = c>>7 is constant within a fragment, d = c&127.
      const int kbase = ki * 32;
      union { v16h h; unsigned u[8]; } af;
      const _Float16* abase =
          &Xp[(mt * 16 + m + (kbase >> 7)) * ND + (kbase & 127)];
#pragma unroll
      for (int j = 0; j < 8; ++j) {
        int off = (half << 4) + (j << 1);
        // element index is even -> byte address is 4-aligned -> b32 LDS load
        af.u[j] = *reinterpret_cast<const unsigned*>(abase + off);
      }
      c = __builtin_amdgcn_wmma_f32_16x16x32_f16(
          /*neg_a=*/false, af.h, /*neg_b=*/false, bf,
          /*c_mod=*/(short)0, c, /*reuse_a=*/false, /*reuse_b=*/false);
    }
    // Epilogue: bias + ReLU; C/D layout: VGPR v -> row (v + 8*half), col n.
#pragma unroll
    for (int v = 0; v < 8; ++v) {
      int row = mt * 16 + v + (half << 3);
      float y = fmaxf(c[v] + bn, 0.0f);
      if constexpr (MEAN) {
        if (row < L) acc += y;
      } else {
        if (row < L) out[outBase + row * ND + n] = y;
      }
    }
  }

  if constexpr (MEAN) {
    // lanes l and l+16 hold the same column n -> combine halves, then write.
    acc += __shfl_xor(acc, 16, 32);
    if (half == 0) out[outBase + n] = acc * (1.0f / L);
  }
}

// ---------------------------------------------------------------------------
// Co-attention + hard routing (argmax approximation of gumbel-hard).
// S = u_sum[b] @ M_co[p] @ i_sum[b]^T ; sel_u = argmax_u max_v S,
// sel_i = argmax_v max_u S.  grid = B, 128 threads.
// ---------------------------------------------------------------------------
__global__ void coatt_select(const float* __restrict__ u_sum,
                             const float* __restrict__ i_sum,
                             const float* __restrict__ M_co,
                             int* __restrict__ sel_u,
                             int* __restrict__ sel_i) {
  __shared__ float Us[NR * ND], Is[NR * ND], T[NR * ND], S[NR * NR];
  int b = blockIdx.x, tid = threadIdx.x;
  for (int i = tid; i < NR * ND; i += 128) {
    Us[i] = u_sum[b * NR * ND + i];
    Is[i] = i_sum[b * NR * ND + i];
  }
  __syncthreads();
  for (int p = 0; p < NP; ++p) {
    const float* M = M_co + p * ND * ND;
    for (int i = tid; i < NR * ND; i += 128) {   // T = Us @ M
      int u = i >> 7, e = i & 127;
      float s = 0.0f;
      for (int d = 0; d < ND; ++d) s += Us[u * ND + d] * M[d * ND + e];
      T[i] = s;
    }
    __syncthreads();
    for (int i = tid; i < NR * NR; i += 128) {   // S = T @ Is^T
      int u = i / NR, v = i % NR;
      float s = 0.0f;
      for (int e = 0; e < ND; ++e) s += T[u * ND + e] * Is[v * ND + e];
      S[i] = s;
    }
    __syncthreads();
    if (tid == 0) {
      int bu = 0, bv = 0;
      float mu = -1e30f, mv = -1e30f;
      for (int u = 0; u < NR; ++u) {
        float mrow = -1e30f;
        for (int v = 0; v < NR; ++v) mrow = fmaxf(mrow, S[u * NR + v]);
        if (mrow > mu) { mu = mrow; bu = u; }
      }
      for (int v = 0; v < NR; ++v) {
        float mcol = -1e30f;
        for (int u = 0; u < NR; ++u) mcol = fmaxf(mcol, S[u * NR + v]);
        if (mcol > mv) { mv = mcol; bv = v; }
      }
      sel_u[p * NB + b] = bu;
      sel_i[p * NB + b] = bv;
    }
    __syncthreads();
  }
}

// ---------------------------------------------------------------------------
// Attention pooling for one p, one side.  grid = B, 128 threads (thread = d).
//   q = [item_asp ; svec] @ Wqv + bqv       [A, D]
//   s = softmax( (q . x_l) / 0.5 , over l ) [A, L]
//   fea[b,a,:] += sum_l s[a,l] * x_l
// ---------------------------------------------------------------------------
__global__ void att_pool(const int* __restrict__ iids,
                         const float* __restrict__ aspect_emb,
                         const float* __restrict__ sum_tab,
                         const int* __restrict__ sel,
                         const float* __restrict__ words,
                         const float* __restrict__ Wqv,
                         const float* __restrict__ bqv,
                         float* __restrict__ fea) {
  __shared__ float q[NA * ND];
  __shared__ float sc[NA * NRL];
  __shared__ float asp[NA * ND];
  __shared__ float sv[ND];
  int b = blockIdx.x, d = threadIdx.x;
  int iid = iids[b];
  for (int i = d; i < NA * ND; i += 128) asp[i] = aspect_emb[(long)iid * NA * ND + i];
  sv[d] = sum_tab[(b * NR + sel[b]) * ND + d];
  __syncthreads();

  float svpart = 0.0f;   // svec half of q is shared across aspects
  for (int j = 0; j < ND; ++j) svpart += sv[j] * Wqv[(ND + j) * ND + d];
  for (int a = 0; a < NA; ++a) {
    float s = bqv[d] + svpart;
    for (int j = 0; j < ND; ++j) s += asp[a * ND + j] * Wqv[j * ND + d];
    q[a * ND + d] = s;
  }
  __syncthreads();

  const float* X = words + (long)b * NRL * ND;
  for (int l = d; l < NRL; l += 128) {
    float dot[NA] = {0, 0, 0, 0, 0};
    for (int dd = 0; dd < ND; ++dd) {
      float x = X[l * ND + dd];
#pragma unroll
      for (int a = 0; a < NA; ++a) dot[a] += q[a * ND + dd] * x;
    }
    for (int a = 0; a < NA; ++a) sc[a * NRL + l] = dot[a] * 2.0f;  // / 0.5
  }
  __syncthreads();

  if (d < NA) {  // softmax over l for aspect d
    float m = -1e30f;
    for (int l = 0; l < NRL; ++l) m = fmaxf(m, sc[d * NRL + l]);
    float s = 0.0f;
    for (int l = 0; l < NRL; ++l) {
      float e = __expf(sc[d * NRL + l] - m);
      sc[d * NRL + l] = e;
      s += e;
    }
    float inv = 1.0f / s;
    for (int l = 0; l < NRL; ++l) sc[d * NRL + l] *= inv;
  }
  __syncthreads();

  float f[NA] = {0, 0, 0, 0, 0};
  for (int l = 0; l < NRL; ++l) {
    float x = X[l * ND + d];
#pragma unroll
    for (int a = 0; a < NA; ++a) f[a] += sc[a * NRL + l] * x;
  }
  for (int a = 0; a < NA; ++a) fea[(long)b * NA * ND + a * ND + d] += f[a];
}

// ---------------------------------------------------------------------------
// Final: stack(id_emb[ids], fea_flat @ W_fc + b_fc).  grid = B, 64 threads.
// out layout: u_final [B,2,32] at offset 0, i_final [B,2,32] at offset 8192.
// ---------------------------------------------------------------------------
__global__ void final_fc(const int* __restrict__ uids, const int* __restrict__ iids,
                         const float* __restrict__ uid_emb,
                         const float* __restrict__ iid_emb,
                         const float* __restrict__ u_fea,
                         const float* __restrict__ i_fea,
                         const float* __restrict__ W_ufc, const float* __restrict__ b_ufc,
                         const float* __restrict__ W_ifc, const float* __restrict__ b_ifc,
                         float* __restrict__ out) {
  int b = blockIdx.x, tid = threadIdx.x;
  int side = tid >> 5, j = tid & 31;
  const float* fea = side ? i_fea : u_fea;
  const float* W   = side ? W_ifc : W_ufc;
  const float* bb  = side ? b_ifc : b_ufc;
  const float* ide = side ? iid_emb : uid_emb;
  int id = side ? iids[b] : uids[b];
  float s = bb[j];
  for (int t = 0; t < NA * ND; ++t) s += fea[(long)b * NA * ND + t] * W[t * NID + j];
  float* o = out + (long)side * (NB * 2 * NID) + b * 2 * NID;
  o[j] = ide[(long)id * NID + j];
  o[NID + j] = s;
}

// ---------------------------------------------------------------------------
extern "C" void kernel_launch(void* const* d_in, const int* in_sizes, int n_in,
                              void* d_out, int out_size, void* d_ws, size_t ws_size,
                              hipStream_t stream) {
  (void)in_sizes; (void)n_in; (void)out_size; (void)ws_size;

  const int*   user_reviews   = (const int*)  d_in[0];
  const int*   item_reviews   = (const int*)  d_in[1];
  const int*   uids           = (const int*)  d_in[2];
  const int*   iids           = (const int*)  d_in[3];
  const int*   user_summaries = (const int*)  d_in[4];
  const int*   item_summaries = (const int*)  d_in[5];
  const float* user_word_emb  = (const float*)d_in[6];
  const float* item_word_emb  = (const float*)d_in[7];
  const float* item_aspect    = (const float*)d_in[8];
  const float* uid_emb        = (const float*)d_in[9];
  const float* iid_emb        = (const float*)d_in[10];
  const float* W_ucnn         = (const float*)d_in[11];
  const float* b_ucnn         = (const float*)d_in[12];
  const float* W_icnn         = (const float*)d_in[13];
  const float* b_icnn         = (const float*)d_in[14];
  const float* M_co           = (const float*)d_in[15];
  const float* W_ufc          = (const float*)d_in[16];
  const float* b_ufc          = (const float*)d_in[17];
  const float* W_ifc          = (const float*)d_in[18];
  const float* b_ifc          = (const float*)d_in[19];
  const float* W_uqv          = (const float*)d_in[20];
  const float* b_uqv          = (const float*)d_in[21];
  const float* W_iqv          = (const float*)d_in[22];
  const float* b_iqv          = (const float*)d_in[23];
  float* out = (float*)d_out;

  // ---- workspace carve-up (256B aligned) ----
  size_t off = 0;
  auto carve = [&](size_t bytes) -> void* {
    void* p = (char*)d_ws + off;
    off += (bytes + 255) & ~(size_t)255;
    return p;
  };
  _Float16* Wpack_u = (_Float16*)carve(KDIM * ND * sizeof(_Float16)); // 96 KB
  _Float16* Wpack_i = (_Float16*)carve(KDIM * ND * sizeof(_Float16));
  float* u_sum  = (float*)carve((size_t)NB * NR * ND * sizeof(float)); // 640 KB
  float* i_sum  = (float*)carve((size_t)NB * NR * ND * sizeof(float));
  int*   sel_u  = (int*)  carve((size_t)NP * NB * sizeof(int));
  int*   sel_i  = (int*)  carve((size_t)NP * NB * sizeof(int));
  float* u_fea  = (float*)carve((size_t)NB * NA * ND * sizeof(float)); // 320 KB
  float* i_fea  = (float*)carve((size_t)NB * NA * ND * sizeof(float));
  float* u_words = (float*)carve((size_t)NB * NRL * ND * sizeof(float)); // 12.5 MB
  float* i_words = (float*)carve((size_t)NB * NRL * ND * sizeof(float));

  // 1) pack conv weights into WMMA B-fragment layout (f32 -> f16)
  pack_weights<<<(2 * KDIM * ND + 255) / 256, 256, 0, stream>>>(
      W_ucnn, W_icnn, Wpack_u, Wpack_i);

  // 2) summary convs (+ mean over L) -> u_sum/i_sum  [B*R, 128]
  conv_wmma<NSL, 2, true><<<NB * NR, 256, 0, stream>>>(
      user_summaries, nullptr, NR, user_word_emb, Wpack_u, b_ucnn, u_sum);
  conv_wmma<NSL, 2, true><<<NB * NR, 256, 0, stream>>>(
      item_summaries, nullptr, NR, item_word_emb, Wpack_i, b_icnn, i_sum);

  // 3) co-attention routing (argmax approximation of gumbel-hard)
  coatt_select<<<NB, 128, 0, stream>>>(u_sum, i_sum, M_co, sel_u, sel_i);

  // 4) zero feature accumulators
  hipMemsetAsync(u_fea, 0, (size_t)NB * NA * ND * sizeof(float), stream);
  hipMemsetAsync(i_fea, 0, (size_t)NB * NA * ND * sizeof(float), stream);

  // 5) per-head: conv the SELECTED review only, then attention-pool
  for (int p = 0; p < NP; ++p) {
    conv_wmma<NRL, 13, false><<<NB, 256, 0, stream>>>(
        user_reviews, sel_u + p * NB, NR, user_word_emb, Wpack_u, b_ucnn, u_words);
    conv_wmma<NRL, 13, false><<<NB, 256, 0, stream>>>(
        item_reviews, sel_i + p * NB, NR, item_word_emb, Wpack_i, b_icnn, i_words);
    att_pool<<<NB, 128, 0, stream>>>(
        iids, item_aspect, u_sum, sel_u + p * NB, u_words, W_uqv, b_uqv, u_fea);
    att_pool<<<NB, 128, 0, stream>>>(
        iids, item_aspect, i_sum, sel_i + p * NB, i_words, W_iqv, b_iqv, i_fea);
  }

  // 6) final FC + id-embedding stack
  final_fc<<<NB, 64, 0, stream>>>(uids, iids, uid_emb, iid_emb, u_fea, i_fea,
                                  W_ufc, b_ufc, W_ifc, b_ifc, out);
}
